// GATLayer_78451872628890
// MI455X (gfx1250) — matrix operand order repaired and checked by reading, not compiled, not run on hardware
//
#include <hip/hip_runtime.h>

typedef __attribute__((ext_vector_type(2)))  float    v2f;
typedef __attribute__((ext_vector_type(4)))  float    v4f;
typedef __attribute__((ext_vector_type(8)))  float    v8f;
typedef __attribute__((ext_vector_type(16))) _Float16 v16h;
typedef __attribute__((ext_vector_type(4)))  int      v4i;

#define N_NODES 4096
#define DIN     128
#define DOUT    64
#define NHEAD   4
#define ALPHA   0.2f

__device__ __forceinline__ float leakyf(float x) { return x > 0.f ? x : ALPHA * x; }

// ---------------------------------------------------------------------------
// Kernel A: Wh[h][n][o] = X[n][:] @ W[h][:][o]   (exact fp32 via WMMA 16x16x4)
// Also writes a transposed f16 copy Wht[h][o][n] used as WMMA B-matrix later.
// One wave32 per (16-row tile, head).
// ---------------------------------------------------------------------------
__global__ __launch_bounds__(32) void gat_wh_kernel(
    const float* __restrict__ X, const float* __restrict__ W,
    float* __restrict__ Wh, _Float16* __restrict__ Wht)
{
    const int rt      = blockIdx.x >> 2;
    const int h       = blockIdx.x & 3;
    const int rowbase = rt * 16;
    const int lane    = threadIdx.x & 31;
    const int m       = lane & 15;   // row within tile (A) / column (B,C)
    const int g       = lane >> 4;   // half-wave group

    const float* Xr = X + (size_t)(rowbase + m) * DIN;
    const float* Wb = W + (size_t)h * DIN * DOUT;

    v8f c[4] = {};   // four 16x16 output tiles -> 16 x 64
    for (int kk = 0; kk < DIN; kk += 4) {
        const int ka = kk + 2 * g;                   // A layout: v0=K(2g), v1=K(2g+1)
        v2f a = *(const v2f*)(Xr + ka);
#pragma unroll
        for (int t = 0; t < 4; ++t) {
            const int n = m + 16 * t;
            v2f b;                                   // B layout mirrors A over K
            b.x = Wb[ka * DOUT + n];
            b.y = Wb[(ka + 1) * DOUT + n];
            c[t] = __builtin_amdgcn_wmma_f32_16x16x4_f32(
                false, a, false, b, (short)0, c[t], false, false);
        }
    }
#pragma unroll
    for (int t = 0; t < 4; ++t) {
        const int n = m + 16 * t;
#pragma unroll
        for (int v = 0; v < 8; ++v) {                // C layout: vgpr v -> row v+8g
            const int row = rowbase + v + 8 * g;
            const float val = c[t][v];
            Wh [(size_t)(h * N_NODES + row) * DOUT + n]   = val;
            Wht[(size_t)(h * DOUT + n) * N_NODES + row]   = (_Float16)val;
        }
    }
}

// ---------------------------------------------------------------------------
// Kernel B: s[h][i] = Wh[h][i][:] . a_src[h][:],  d likewise. Trivial.
// ---------------------------------------------------------------------------
__global__ __launch_bounds__(256) void gat_score_kernel(
    const float* __restrict__ Wh, const float* __restrict__ a_src,
    const float* __restrict__ a_dst, float* __restrict__ s, float* __restrict__ d)
{
    const int t = blockIdx.x * blockDim.x + threadIdx.x;
    if (t >= NHEAD * N_NODES) return;
    const int h = t / N_NODES;
    const float* w  = Wh + (size_t)t * DOUT;
    const float* as = a_src + h * DOUT;
    const float* ad = a_dst + h * DOUT;
    float accs = 0.f, accd = 0.f;
#pragma unroll
    for (int o = 0; o < DOUT; o += 4) {
        v4f wv = *(const v4f*)(w + o);
        v4f sv = *(const v4f*)(as + o);
        v4f dv = *(const v4f*)(ad + o);
        accs += wv.x * sv.x + wv.y * sv.y + wv.z * sv.z + wv.w * sv.w;
        accd += wv.x * dv.x + wv.y * dv.y + wv.z * dv.z + wv.w * dv.w;
    }
    s[t] = accs;
    d[t] = accd;
}

// ---------------------------------------------------------------------------
// Kernel C: fused masked-softmax attention + aggregation.
// One wave32 per (16-row tile, head).
// Pass 1: rowmax_i = leaky(s_i + max_{j in nbr(i)} d_j)   (leaky is monotone)
// Pass 2: P_ij = exp(leaky(s_i+d_j) - rowmax_i) (masked); accumulate rowsum
//         and  C += P(16x32,f16) @ WhT-block(32x16,f16)  via v_wmma_f32_16x16x32_f16.
// Finally divide by rowsum (every row has a self-loop -> rowsum > 0).
// ---------------------------------------------------------------------------
__global__ __launch_bounds__(32) void gat_attn_kernel(
    const int* __restrict__ adj, const float* __restrict__ s,
    const float* __restrict__ dvec, const _Float16* __restrict__ Wht,
    float* __restrict__ out)
{
    const int rt      = blockIdx.x >> 2;
    const int h       = blockIdx.x & 3;
    const int rowbase = rt * 16;
    const int lane    = threadIdx.x & 31;
    const int m       = lane & 15;
    const int g       = lane >> 4;
    const int row     = rowbase + m;

    const float  sm   = s[h * N_NODES + row];
    const float* dh   = dvec + (size_t)h * N_NODES;
    const int*   adjr = adj + (size_t)row * N_NODES;

    // ---- pass 1: masked max of d over this row (lane handles half the cols)
    float mx = -3.0e38f;
    const int c0 = g * (N_NODES / 2);
    for (int c = 0; c < N_NODES / 2; c += 16) {
        __builtin_prefetch(adjr + c0 + c + 256, 0, 0);   // global_prefetch_b8
#pragma unroll
        for (int q = 0; q < 16; q += 4) {
            v4i av = *(const v4i*)(adjr + c0 + c + q);
            v4f dv = *(const v4f*)(dh  + c0 + c + q);
            if (av.x > 0) mx = fmaxf(mx, dv.x);
            if (av.y > 0) mx = fmaxf(mx, dv.y);
            if (av.z > 0) mx = fmaxf(mx, dv.z);
            if (av.w > 0) mx = fmaxf(mx, dv.w);
        }
    }
    mx = fmaxf(mx, __shfl_xor(mx, 16, 32));   // combine the two column halves
    mx = fmaxf(mx, dh[row]);                  // self-loop always present
    const float rm = leakyf(sm + mx);         // softmax row max (monotone leaky)

    // ---- pass 2: fused exp + rowsum + WMMA aggregation
    v8f   c[4] = {};
    float rs   = 0.f;
    for (int cb = 0; cb < N_NODES; cb += 32) {
        __builtin_prefetch(adjr + cb + 512, 0, 0);
        const int j0 = cb + 8 * g;        // A-matrix K groups for this lane:
        const int j1 = cb + 16 + 8 * g;   // [8g,8g+8) and [16+8g,16+8g+8)
        v4i a0 = *(const v4i*)(adjr + j0);
        v4i a1 = *(const v4i*)(adjr + j0 + 4);
        v4i a2 = *(const v4i*)(adjr + j1);
        v4i a3 = *(const v4i*)(adjr + j1 + 4);
        v4f d0 = *(const v4f*)(dh + j0);
        v4f d1 = *(const v4f*)(dh + j0 + 4);
        v4f d2 = *(const v4f*)(dh + j1);
        v4f d3 = *(const v4f*)(dh + j1 + 4);

        const int   adjv[16] = {a0.x,a0.y,a0.z,a0.w, a1.x,a1.y,a1.z,a1.w,
                                a2.x,a2.y,a2.z,a2.w, a3.x,a3.y,a3.z,a3.w};
        const float dval[16] = {d0.x,d0.y,d0.z,d0.w, d1.x,d1.y,d1.z,d1.w,
                                d2.x,d2.y,d2.z,d2.w, d3.x,d3.y,d3.z,d3.w};

        v16h A;
#pragma unroll
        for (int e = 0; e < 16; ++e) {
            const int  j   = (e < 8 ? j0 + e : j1 + (e - 8));
            const float ev = leakyf(sm + dval[e]);
            const bool msk = (adjv[e] > 0) || (j == row);
            const float pv = msk ? __expf(ev - rm) : 0.f;
            rs  += pv;
            A[e] = (_Float16)pv;
        }

        // B: Wht[h][n][cb+8g .. +7] and [cb+16+8g .. +7] -> two 16B loads per lane
        const _Float16* bbase =
            Wht + (size_t)(h * DOUT) * N_NODES + cb + 8 * g;
#pragma unroll
        for (int t = 0; t < 4; ++t) {
            const int n = m + 16 * t;
            v4i blo = *(const v4i*)(bbase + (size_t)n * N_NODES);
            v4i bhi = *(const v4i*)(bbase + (size_t)n * N_NODES + 16);
            v16h B = (v16h)__builtin_shufflevector(blo, bhi, 0, 1, 2, 3, 4, 5, 6, 7);
            c[t] = __builtin_amdgcn_wmma_f32_16x16x32_f16(
                false, A, false, B, (short)0, c[t], false, false);
        }
    }

    // combine per-half rowsums: lanes l and l+16 hold disjoint K subsets of row m
    rs += __shfl_xor(rs, 16, 32);

    float inv[8];
#pragma unroll
    for (int v = 0; v < 8; ++v)
        inv[v] = 1.0f / __shfl(rs, v + 8 * g, 32);   // row v+8g is held by lane v+8g

#pragma unroll
    for (int t = 0; t < 4; ++t) {
        const int n = m + 16 * t;
#pragma unroll
        for (int v = 0; v < 8; ++v) {
            const int r = rowbase + v + 8 * g;
            out[(size_t)r * (NHEAD * DOUT) + h * DOUT + n] = c[t][v] * inv[v];
        }
    }
}

// ---------------------------------------------------------------------------
extern "C" void kernel_launch(void* const* d_in, const int* in_sizes, int n_in,
                              void* d_out, int out_size, void* d_ws, size_t ws_size,
                              hipStream_t stream) {
    const float* X     = (const float*)d_in[0];
    const int*   adj   = (const int*)  d_in[1];
    const float* W     = (const float*)d_in[2];
    const float* a_src = (const float*)d_in[3];
    const float* a_dst = (const float*)d_in[4];
    float*       out   = (float*)d_out;

    char* ws = (char*)d_ws;
    float*    Wh  = (float*)ws;                              // 4 MB  f32 [H][N][DOUT]
    _Float16* Wht = (_Float16*)(ws + 4u * 1024u * 1024u);    // 2 MB  f16 [H][DOUT][N]
    float*    s   = (float*)(ws + 6u * 1024u * 1024u);       // 64 KB
    float*    d   = (float*)(ws + 6u * 1024u * 1024u + 64u * 1024u); // 64 KB

    gat_wh_kernel   <<<N_NODES / 16 * NHEAD, 32, 0, stream>>>(X, W, Wh, Wht);
    gat_score_kernel<<<(NHEAD * N_NODES + 255) / 256, 256, 0, stream>>>(Wh, a_src, a_dst, s, d);
    gat_attn_kernel <<<N_NODES / 16 * NHEAD, 32, 0, stream>>>(adj, s, d, Wht, out);
}
